// NodeGNN_38654705664132
// MI455X (gfx1250) — matrix operand order repaired and checked by reading, not compiled, not run on hardware
//
#include <hip/hip_runtime.h>
#include <hip/hip_bf16.h>

typedef __attribute__((ext_vector_type(16))) _Float16 v16h;
typedef __attribute__((ext_vector_type(8)))  _Float16 v8h;
typedef __attribute__((ext_vector_type(8)))  float    v8f;

#define IN_DIM 128
#define H1DIM  256
#define HID    16
#define NEG_SLOPE 0.01f

__device__ __forceinline__ float lrelu(float v) {
    return v >= 0.0f ? v : NEG_SLOPE * v;
}

// ---------------------------------------------------------------------------
// Weight pre-swizzle into WMMA B-fragment layout (f16).
// B-frag (32x16, 16x16x32_f16): lane l holds col = l&15,
//   element e (0..15) = B[k][col] with k = 32*kc + 16*(l>=16) + e.
// Storage: frag f = c*KC + kc, flat index ((f*32)+l)*16 + e  (32B per lane).
// ---------------------------------------------------------------------------
__global__ void prep_w1_kernel(const float* __restrict__ W1, _Float16* __restrict__ w1f) {
    int gid = blockIdx.x * blockDim.x + threadIdx.x;
    if (gid >= 16 * 4 * 32 * 16) return;              // 16 coltiles * 4 kchunks
    int e  = gid & 15;
    int l  = (gid >> 4) & 31;
    int f  = gid >> 9;
    int c  = f >> 2;          // col tile (0..15)
    int kc = f & 3;           // k chunk  (0..3)
    int col = c * 16 + (l & 15);
    int k   = kc * 32 + (l >> 4) * 16 + e;
    w1f[gid] = (_Float16)W1[k * H1DIM + col];
}

__global__ void prep_w2_kernel(const float* __restrict__ W2, _Float16* __restrict__ w2f) {
    int gid = blockIdx.x * blockDim.x + threadIdx.x;
    if (gid >= 8 * 32 * 16) return;                   // 8 kchunks, one col tile
    int e  = gid & 15;
    int l  = (gid >> 4) & 31;
    int kc = gid >> 9;        // 0..7
    int col = l & 15;
    int k   = kc * 32 + (l >> 4) * 16 + e;
    w2f[gid] = (_Float16)W2[k * HID + col];
}

// ---------------------------------------------------------------------------
// Fused MLP: h = lrelu(lrelu(x@W1+b1)@W2+b2), one 16-node tile per workgroup.
// 4 waves: stage1 each wave does 4 col-tiles of the 256-wide h1 (16 WMMA),
// stage2 each wave does K=64 slice of h1@W2 (2 WMMA), reduced in LDS.
// h1 never touches HBM.
// ---------------------------------------------------------------------------
__global__ __launch_bounds__(128) void mlp_kernel(
    const float*   __restrict__ x,
    const _Float16* __restrict__ w1f,
    const float*   __restrict__ b1,
    const _Float16* __restrict__ w2f,
    const float*   __restrict__ b2,
    float*         __restrict__ h)
{
    __shared__ _Float16 xt[16 * IN_DIM];    // 4 KB  x tile (f16)
    __shared__ _Float16 h1t[16 * H1DIM];    // 8 KB  h1 tile (f16)
    __shared__ float    part[4 * 256];      // 4 KB  stage-2 partials

    const int t     = threadIdx.x;
    const int node0 = blockIdx.x * 16;

    // ---- stage 0: cooperative load + f32->f16 of the x tile -------------
    {
        int row = t >> 3;            // 0..15
        int c0  = (t & 7) * 16;      // 0,16,...,112
        const float* src = x + (size_t)(node0 + row) * IN_DIM + c0;
        #pragma unroll
        for (int i = 0; i < 16; ++i)
            xt[row * IN_DIM + c0 + i] = (_Float16)src[i];
    }
    __syncthreads();

    const int wave = t >> 5;
    const int lane = t & 31;
    const int row  = lane & 15;
    const int hi   = lane >> 4;      // 0 | 1

    // ---- stage 1: x(16x128) @ W1(128x256), this wave's 4 col-tiles ------
    v8f acc[4];
    #pragma unroll
    for (int cc = 0; cc < 4; ++cc) {
        v8f z = {0.f,0.f,0.f,0.f,0.f,0.f,0.f,0.f};
        acc[cc] = z;
    }

    #pragma unroll
    for (int kc = 0; kc < 4; ++kc) {
        const int kb = kc * 32 + hi * 16;
        v8h alo = *(const v8h*)(xt + row * IN_DIM + kb);
        v8h ahi = *(const v8h*)(xt + row * IN_DIM + kb + 8);
        v16h a  = __builtin_shufflevector(alo, ahi,
                    0,1,2,3,4,5,6,7,8,9,10,11,12,13,14,15);
        #pragma unroll
        for (int cc = 0; cc < 4; ++cc) {
            const int c = wave * 4 + cc;
            v16h b = *(const v16h*)(w1f + ((size_t)((c * 4 + kc) * 32 + lane)) * 16);
            acc[cc] = __builtin_amdgcn_wmma_f32_16x16x32_f16(
                false, a, false, b, (short)0, acc[cc], false, false);
        }
    }

    // bias + leaky-relu + f16 -> h1 tile in LDS
    #pragma unroll
    for (int cc = 0; cc < 4; ++cc) {
        const int col = (wave * 4 + cc) * 16 + row;
        const float bias = b1[col];
        #pragma unroll
        for (int r = 0; r < 8; ++r) {
            const int m = r + hi * 8;
            h1t[m * H1DIM + col] = (_Float16)lrelu(acc[cc][r] + bias);
        }
    }
    __syncthreads();

    // ---- stage 2: h1(16x256) @ W2(256x16); wave handles K-chunks 2w,2w+1
    v8f acc2 = {0.f,0.f,0.f,0.f,0.f,0.f,0.f,0.f};
    #pragma unroll
    for (int kk = 0; kk < 2; ++kk) {
        const int kc = wave * 2 + kk;
        const int kb = kc * 32 + hi * 16;
        v8h alo = *(const v8h*)(h1t + row * H1DIM + kb);
        v8h ahi = *(const v8h*)(h1t + row * H1DIM + kb + 8);
        v16h a  = __builtin_shufflevector(alo, ahi,
                    0,1,2,3,4,5,6,7,8,9,10,11,12,13,14,15);
        v16h b  = *(const v16h*)(w2f + (size_t)(kc * 32 + lane) * 16);
        acc2 = __builtin_amdgcn_wmma_f32_16x16x32_f16(
            false, a, false, b, (short)0, acc2, false, false);
    }
    #pragma unroll
    for (int r = 0; r < 8; ++r) {
        const int m = r + hi * 8;
        part[wave * 256 + m * 16 + row] = acc2[r];
    }
    __syncthreads();

    // ---- stage 3: reduce 4 wave-partials, bias, lrelu, store h ----------
    #pragma unroll
    for (int i = 0; i < 2; ++i) {
        const int idx = t * 2 + i;               // 0..255
        const int m = idx >> 4, col = idx & 15;
        float v = part[idx] + part[256 + idx] + part[512 + idx] + part[768 + idx]
                + b2[col];
        h[(size_t)(node0 + m) * HID + col] = lrelu(v);
    }
}

// ---------------------------------------------------------------------------
// Graph kernels (all L2-resident node state on MI455X's 192MB L2)
// ---------------------------------------------------------------------------
__global__ void fill_kernel(float* __restrict__ p, float v, long n) {
    long gid = (long)blockIdx.x * blockDim.x + threadIdx.x;
    if (gid < n) p[gid] = v;
}

__global__ void deg_kernel(const int* __restrict__ dst, float* __restrict__ deg, int E) {
    int e = blockIdx.x * blockDim.x + threadIdx.x;
    if (e < E) atomicAdd(&deg[dst[e]], 1.0f);
}

__global__ void dinv_kernel(const float* __restrict__ deg, float* __restrict__ dinv, int N) {
    int n = blockIdx.x * blockDim.x + threadIdx.x;
    if (n < N) {
        float d = deg[n];
        dinv[n] = d > 0.0f ? rsqrtf(d) : 0.0f;
    }
}

// hw = h @ W  (16x16 per node; memory-bound, fp32 VALU)
__global__ void hw_kernel(const float* __restrict__ h, const float* __restrict__ W,
                          float* __restrict__ hw, long n16) {
    long gid = (long)blockIdx.x * blockDim.x + threadIdx.x;
    if (gid >= n16) return;
    long n = gid >> 4;
    int  j = gid & 15;
    const float* hr = h + n * HID;
    float s = 0.0f;
    #pragma unroll
    for (int k = 0; k < HID; ++k) s += hr[k] * W[k * HID + j];
    hw[gid] = s;
}

// edge scatter: agg[dst] += hw[src] * dinv[src]*dinv[dst]   (thread per edge-feature)
__global__ void scatter_kernel(const int* __restrict__ src, const int* __restrict__ dst,
                               const float* __restrict__ dinv, const float* __restrict__ hw,
                               float* __restrict__ agg, long total) {
    long gid = (long)blockIdx.x * blockDim.x + threadIdx.x;
    if (gid >= total) return;
    long e = gid >> 4;
    int  j = gid & 15;
    int  s = src[e];
    int  d = dst[e];
    float norm = dinv[s] * dinv[d];
    atomicAdd(&agg[(size_t)d * HID + j], hw[(size_t)s * HID + j] * norm);
}

// h_out = lrelu(agg + hw*dinv^2 (self loop) + b)
__global__ void finalize_kernel(const float* __restrict__ agg, const float* __restrict__ hw,
                                const float* __restrict__ dinv, const float* __restrict__ b,
                                float* __restrict__ hout, long n16) {
    long gid = (long)blockIdx.x * blockDim.x + threadIdx.x;
    if (gid >= n16) return;
    long n = gid >> 4;
    int  j = gid & 15;
    float di = dinv[n];
    hout[gid] = lrelu(agg[gid] + hw[gid] * di * di + b[j]);
}

// out[n] = sum_j (h@pw + pb)[j] = dot(h[n], rowsum(pw)) + sum(pb)
__global__ void out_kernel(const float* __restrict__ h, const float* __restrict__ pw,
                           const float* __restrict__ pb, float* __restrict__ out, int N) {
    int n = blockIdx.x * blockDim.x + threadIdx.x;
    if (n >= N) return;
    float s = pb[0] + pb[1];
    const float* hr = h + (size_t)n * HID;
    #pragma unroll
    for (int k = 0; k < HID; ++k) s += hr[k] * (pw[k * 2] + pw[k * 2 + 1]);
    out[n] = s;
}

// ---------------------------------------------------------------------------
extern "C" void kernel_launch(void* const* d_in, const int* in_sizes, int n_in,
                              void* d_out, int out_size, void* d_ws, size_t ws_size,
                              hipStream_t stream) {
    const float* x   = (const float*)d_in[0];
    const float* W1  = (const float*)d_in[1];
    const float* b1  = (const float*)d_in[2];
    const float* W2  = (const float*)d_in[3];
    const float* b2  = (const float*)d_in[4];
    const float* cw0 = (const float*)d_in[5];
    const float* cb0 = (const float*)d_in[6];
    const float* cw1 = (const float*)d_in[7];
    const float* cb1 = (const float*)d_in[8];
    const float* pw  = (const float*)d_in[9];
    const float* pb  = (const float*)d_in[10];
    const int*   ei  = (const int*)d_in[11];

    const int N = in_sizes[0] / IN_DIM;     // 200000
    const int E = in_sizes[11] / 2;         // 6400000
    const int* src = ei;
    const int* dst = ei + E;

    // workspace layout (MB-aligned regions)
    char* ws = (char*)d_ws;
    _Float16* w1f  = (_Float16*)(ws + 0);                 // 64 KB
    _Float16* w2f  = (_Float16*)(ws + (1u << 20));        //  8 KB
    float*    h    = (float*)(ws + (2u << 20));           // N*16*4 = 12.8 MB
    float*    hw   = (float*)(ws + (18u << 20));          // 12.8 MB
    float*    agg  = (float*)(ws + (34u << 20));          // 12.8 MB
    float*    deg  = (float*)(ws + (50u << 20));          // 0.8 MB
    float*    dinv = (float*)(ws + (51u << 20));          // 0.8 MB

    float* out_vec = (float*)d_out;                       // [N]
    float* h_final = out_vec + N;                         // [N,16]

    const long n16   = (long)N * HID;
    const long etot  = (long)E * HID;
    const int  B     = 256;
    auto blk = [](long n, int b) { return (unsigned)((n + b - 1) / b); };

    // weight swizzle into WMMA fragment layout
    prep_w1_kernel<<<blk(16 * 4 * 32 * 16, B), B, 0, stream>>>(W1, w1f);
    prep_w2_kernel<<<blk(8 * 32 * 16, B), B, 0, stream>>>(W2, w2f);

    // fused WMMA MLP -> h
    mlp_kernel<<<N / 16, 128, 0, stream>>>(x, w1f, b1, w2f, b2, h);

    // degrees (init to 1.0 = self loop) and dinv
    fill_kernel<<<blk(N, B), B, 0, stream>>>(deg, 1.0f, N);
    deg_kernel<<<blk(E, B), B, 0, stream>>>(dst, deg, E);
    dinv_kernel<<<blk(N, B), B, 0, stream>>>(deg, dinv, N);

    // GCN layer 0: h <- lrelu(agg(h@cw0) + cb0)
    hw_kernel<<<blk(n16, B), B, 0, stream>>>(h, cw0, hw, n16);
    fill_kernel<<<blk(n16, B), B, 0, stream>>>(agg, 0.0f, n16);
    scatter_kernel<<<blk(etot, B), B, 0, stream>>>(src, dst, dinv, hw, agg, etot);
    finalize_kernel<<<blk(n16, B), B, 0, stream>>>(agg, hw, dinv, cb0, h, n16);

    // GCN layer 1: h_final <- lrelu(agg(h@cw1) + cb1)   (written into d_out)
    hw_kernel<<<blk(n16, B), B, 0, stream>>>(h, cw1, hw, n16);
    fill_kernel<<<blk(n16, B), B, 0, stream>>>(agg, 0.0f, n16);
    scatter_kernel<<<blk(etot, B), B, 0, stream>>>(src, dst, dinv, hw, agg, etot);
    finalize_kernel<<<blk(n16, B), B, 0, stream>>>(agg, hw, dinv, cb1, h_final, n16);

    // out[n] = rowsum(h_final @ pw + pb)
    out_kernel<<<blk(N, B), B, 0, stream>>>(h_final, pw, pb, out_vec, N);
}
